// Forest_67989332296124
// MI455X (gfx1250) — compile-verified, matrix operand order checked by reading, not run on hardware
//
#include <hip/hip_runtime.h>

typedef __bf16 bf16x16 __attribute__((ext_vector_type(16)));
typedef float  f32x8   __attribute__((ext_vector_type(8)));
typedef unsigned int u32x4 __attribute__((ext_vector_type(4)));

#define LDSW 522  // 512 + 10 pad (row stride ≡ 5 dwords mod 64 -> conflict-free A reads)

__device__ __forceinline__ unsigned short f2bf(float f) {
    unsigned u = __builtin_bit_cast(unsigned, f);
    unsigned r = (u + 0x7FFFu + ((u >> 16) & 1u)) >> 16;   // round-to-nearest-even
    return (unsigned short)r;
}

// ---------------- prep 1: recover idx[t][l] from one-hot feature_mask ----------------
__global__ __launch_bounds__(256) void k_idx(const float* __restrict__ fm, int* __restrict__ idx) {
    int t = blockIdx.x, l = threadIdx.x;
    const float* p = fm + t * 512 * 256 + l;
    int k = 0;
    for (int f = 0; f < 512; ++f)
        if (p[f * 256] > 0.5f) k = f;
    idx[t * 256 + l] = k;
}

// ---------------- prep 2: Weff[t][k][n] = sum_{l: idx==k} W[t][l][n], pre-swizzled ----
// layout: [t][kb 0..15][nt 0..15][lane 0..31][slot 0..15] matching bf16 B 32x16 fragments
__global__ __launch_bounds__(256) void k_weff(const float* __restrict__ W,
                                              const int* __restrict__ idx,
                                              unsigned short* __restrict__ weff) {
    __shared__ int il[256];
    int t = blockIdx.x >> 9;
    il[threadIdx.x] = idx[t * 256 + threadIdx.x];
    __syncthreads();
    int q    = ((blockIdx.x & 511) << 8) | threadIdx.x;     // 0..131071 within tree
    int slot = q & 15;
    int lane = (q >> 4) & 31;
    int nt   = (q >> 9) & 15;
    int kb   = (q >> 13) & 15;
    int v = slot >> 1, h = slot & 1;
    int k = kb * 32 + ((v < 4) ? 2 * v : 16 + 2 * (v - 4)) + ((lane >> 4) << 3) + h;
    int n = (nt << 4) | (lane & 15);
    const float* Wt = W + t * 65536 + n;
    float acc = 0.f;
    for (int l = 0; l < 256; ++l)
        if (il[l] == k) acc += Wt[l * 256];
    weff[t * 131072 + q] = f2bf(acc);
}

// ---------------- main: GEMM(WMMA bf16) + sigmoid + soft routing + pi contraction ----
// 512 blocks x 256 threads; block tile = 64 rows; wave tile = 32 rows x 64 cols
__global__ __launch_bounds__(256) void k_forest(const float* __restrict__ x,
                                                const unsigned short* __restrict__ weff,
                                                const float* __restrict__ bias,
                                                const float* __restrict__ pi,
                                                float* __restrict__ out) {
#if defined(__gfx1250__)
    __shared__ unsigned short xs[64 * LDSW];   // bf16 x tile       66,816 B
    __shared__ float dls[64 * 256];            // decisions d        65,536 B
    __shared__ float pil[2560];                // pi[t] tile         10,240 B
    __shared__ float preds[640];               // per-row class acc   2,560 B

    const int tid = threadIdx.x;
    const int r0  = blockIdx.x * 64;

    // stage x tile f32 -> bf16 (float4-vectorized, coalesced b128); zero preds
    for (int i = tid; i < 64 * 128; i += 256) {
        int row = i >> 7, c4 = i & 127;
        float4 v = *((const float4*)(x + (r0 + row) * 512) + c4);
        unsigned lo = (unsigned)f2bf(v.x) | ((unsigned)f2bf(v.y) << 16);
        unsigned hi = (unsigned)f2bf(v.z) | ((unsigned)f2bf(v.w) << 16);
        unsigned* dst = (unsigned*)(xs + row * LDSW) + (c4 << 1);
        dst[0] = lo;
        dst[1] = hi;
    }
    for (int i = tid; i < 640; i += 256) preds[i] = 0.f;
    __syncthreads();

    const int lane = tid & 31;
    const int wave = tid >> 5;
    const int wm   = wave >> 2;                        // M group: rows wm*32 .. +32
    const int wn   = wave & 3;                         // N group: cols wn*64
    const int mA   = wm * 32 + (lane & 15);            // A-fragment row (tile 0)
    const int kh2  = (lane >> 4) << 2;                 // K-half select (dwords)
    const int mrb  = wm * 32 + ((lane >> 4) << 3);     // C-fragment base row
    const unsigned* arow0 = (const unsigned*)(xs + mA * LDSW);
    const unsigned* arow1 = (const unsigned*)(xs + (mA + 16) * LDSW);

    const int rw  = tid >> 3;                          // routing base row 0..31
    const int sub = tid & 7;                           // routing: 32-leaf subtree

#pragma unroll 1
    for (int t = 0; t < 16; ++t) {
        // stage pi[t] (safe: previous iteration ended with a barrier)
        for (int i = tid; i < 2560; i += 256) pil[i] = pi[t * 2560 + i];

        // ---- GEMM: 32x64 tile per wave, K = 512; B fragment reused for 2 WMMAs ----
        f32x8 acc[8];
#pragma unroll
        for (int j = 0; j < 8; ++j)
#pragma unroll
            for (int r = 0; r < 8; ++r) acc[j][r] = 0.f;

        const u32x4* wq = (const u32x4*)weff + (t << 14) + (lane << 1);
#pragma unroll 2
        for (int kb = 0; kb < 16; ++kb) {
            union { bf16x16 v; unsigned u[8]; } A0, A1;
#pragma unroll
            for (int v = 0; v < 8; ++v) {
                int dw = (kb << 4) + ((v < 4) ? v : v + 4) + kh2;
                A0.u[v] = arow0[dw];
                A1.u[v] = arow1[dw];
            }
#pragma unroll
            for (int j = 0; j < 4; ++j) {
                int nt = (wn << 2) + j;
                union { bf16x16 v; u32x4 q[2]; } Bf;
                const u32x4* bp = wq + ((kb << 4) + nt) * 64;
                Bf.q[0] = bp[0];
                Bf.q[1] = bp[1];
                acc[j] = __builtin_amdgcn_wmma_f32_16x16x32_bf16(
                    false, A0.v, false, Bf.v, (short)0, acc[j], false, false);
                acc[4 + j] = __builtin_amdgcn_wmma_f32_16x16x32_bf16(
                    false, A1.v, false, Bf.v, (short)0, acc[4 + j], false, false);
            }
        }

        // ---- epilogue: bias + sigmoid -> dls ----
#pragma unroll
        for (int j = 0; j < 4; ++j) {
            int ncol = (wn << 6) + (j << 4) + (lane & 15);
            float bb = bias[(t << 8) + ncol];
#pragma unroll
            for (int r = 0; r < 8; ++r) {
                float z0 = acc[j][r] + bb;
                float z1 = acc[4 + j][r] + bb;
                dls[(mrb + r) * 256 + ncol]      = 1.0f / (1.0f + __expf(-z0));
                dls[(mrb + 16 + r) * 256 + ncol] = 1.0f / (1.0f + __expf(-z1));
            }
        }
        __syncthreads();

        // ---- soft routing + pi contraction: 2 rows per thread ----
#pragma unroll 1
        for (int rr = 0; rr < 2; ++rr) {
            const int row = rw + (rr << 5);
            const float* dr = dls + row * 256;

            // depth-3 prefix
            float dv0 = dr[1];
            float pre = (sub & 4) ? (1.0f - dv0) : dv0;
            int node = 2 + ((sub >> 2) & 1);
            float dv1 = dr[node];
            pre *= (sub & 2) ? (1.0f - dv1) : dv1;
            node = 2 * node + ((sub >> 1) & 1);
            float dv2 = dr[node];
            pre *= (sub & 1) ? (1.0f - dv2) : dv2;

            // depth-5 subtree expansion in registers
            float mu[32];
            mu[0] = pre;
            int nb = 8 + sub;
#define EXPAND(Wd)                              \
            _Pragma("unroll")                    \
            for (int i = (Wd) - 1; i >= 0; --i) {\
                float dv = dr[nb + i];           \
                float m_ = mu[i];                \
                mu[2 * i + 1] = m_ * (1.0f - dv);\
                mu[2 * i]     = m_ * dv;         \
            }                                    \
            nb <<= 1;
            EXPAND(1) EXPAND(2) EXPAND(4) EXPAND(8) EXPAND(16)
#undef EXPAND

            // preds partial: (mu + eps) @ pi
            float p[10];
#pragma unroll
            for (int c = 0; c < 10; ++c) p[c] = 0.f;
#pragma unroll
            for (int i = 0; i < 32; ++i) {
                float m_ = mu[i] + 1.1920929e-7f;
                const float* pp = pil + (sub * 32 + i) * 10;
#pragma unroll
                for (int c = 0; c < 10; ++c) p[c] += m_ * pp[c];
            }
            // reduce across the 8 subtree lanes (consecutive lanes in wave32)
#pragma unroll
            for (int c = 0; c < 10; ++c) {
                p[c] += __shfl_xor(p[c], 1);
                p[c] += __shfl_xor(p[c], 2);
                p[c] += __shfl_xor(p[c], 4);
            }
            if (sub == 0) {
#pragma unroll
                for (int c = 0; c < 10; ++c) preds[row * 10 + c] += p[c];
            }
        }
        __syncthreads();
    }

    for (int i = tid; i < 640; i += 256)
        out[r0 * 10 + i] = preds[i] * 0.0625f;   // /16 trees
#endif
}

extern "C" void kernel_launch(void* const* d_in, const int* in_sizes, int n_in,
                              void* d_out, int out_size, void* d_ws, size_t ws_size,
                              hipStream_t stream) {
    const float* x  = (const float*)d_in[0];
    const float* fm = (const float*)d_in[1];
    const float* W  = (const float*)d_in[2];
    const float* b  = (const float*)d_in[3];
    const float* pi = (const float*)d_in[4];
    float* out = (float*)d_out;

    int* idx = (int*)d_ws;                                         // 16 KB
    unsigned short* weff = (unsigned short*)((char*)d_ws + 16384); // 4 MB bf16, swizzled

    k_idx  <<<16,   256, 0, stream>>>(fm, idx);
    k_weff <<<8192, 256, 0, stream>>>(W, idx, weff);
    k_forest<<<512, 256, 0, stream>>>(x, weff, b, pi, out);
}
